// StudentMoE_88751204205084
// MI455X (gfx1250) — matrix operand order repaired and checked by reading, not compiled, not run on hardware
//
#include <hip/hip_runtime.h>
#include <hip/hip_bf16.h>
#include <hip/hip_fp16.h>

// ---------------------------------------------------------------------------
// Model dims (compile-time, from the reference)
// ---------------------------------------------------------------------------
#define D_MODEL 1024
#define NHEAD   16
#define NKVH    4
#define HEADD   64
#define HIDDEN  1024
#define NEXP    8
#define TOPK_   2
#define VOCAB   32000
#define NLAYER  2
#define BATCH   2
#define SEQ     1024
#define NTOK    (BATCH * SEQ)     // 2048
#define NPAIR   (NTOK * TOPK_)    // 4096
#define QDIM    (NHEAD * HEADD)   // 1024
#define KVDIM   (NKVH * HEADD)    // 256

typedef __attribute__((ext_vector_type(16))) _Float16 v16h;
typedef __attribute__((ext_vector_type(8)))  _Float16 v8h;
typedef __attribute__((ext_vector_type(8)))  float    v8f;

// ---------------------------------------------------------------------------
// CDNA5 async global->LDS staging (ASYNCcnt path), with compile-safe fallback
// ---------------------------------------------------------------------------
#if __has_builtin(__builtin_amdgcn_global_load_async_to_lds_b128)
#define USE_ASYNC_LDS 1
#else
#define USE_ASYNC_LDS 0
#endif

typedef int i4v __attribute__((vector_size(16)));
typedef __attribute__((address_space(1))) i4v gi4v;  // global int4
typedef __attribute__((address_space(3))) i4v li4v;  // LDS int4

__device__ __forceinline__ void cp16(const _Float16* g, _Float16* l) {
#if USE_ASYNC_LDS
  __builtin_amdgcn_global_load_async_to_lds_b128((gi4v*)g, (li4v*)l, 0, 0);
#else
  *(int4*)l = *(const int4*)g;
#endif
}

__device__ __forceinline__ void wait_async() {
#if USE_ASYNC_LDS
#if __has_builtin(__builtin_amdgcn_s_wait_asynccnt)
  __builtin_amdgcn_s_wait_asynccnt(0);
#else
  asm volatile("s_wait_asynccnt 0" ::: "memory");
#endif
#endif
}

// ---------------------------------------------------------------------------
// WMMA helpers: D = A(16x32 f16) * B(32x16 f16) + C(16x16 f32), wave32
// ---------------------------------------------------------------------------
__device__ __forceinline__ v8f wmma16(v16h a, v16h b, v8f c) {
  return __builtin_amdgcn_wmma_f32_16x16x32_f16(false, a, false, b, (short)0, c,
                                                false, false);
}

// A fragment (contiguous K order: half selects K 0..15/16..31). LDS tile: [row][k], k-stride 32.
__device__ __forceinline__ v16h frag_a(const _Float16* s, int rowbase, int lane) {
  const int m  = lane & 15;
  const int hf = lane >> 4;
  const _Float16* p = s + (rowbase + m) * 32 + hf * 16;
  v8h lo = *(const v8h*)p;         // K = hf*16 + 0..7
  v8h hi = *(const v8h*)(p + 8);   // K = hf*16 + 8..15
  v16h r;
#pragma unroll
  for (int i = 0; i < 8; ++i) { r[i] = lo[i]; r[i + 8] = hi[i]; }
  return r;
}

// B fragment: lane = output column (n), half selects contiguous K 0..15/16..31.
__device__ __forceinline__ v16h frag_b(const _Float16* s, int colbase, int lane) {
  const int n  = lane & 15;
  const int hf = lane >> 4;
  return *(const v16h*)(s + (colbase + n) * 32 + hf * 16);
}

// ---------------------------------------------------------------------------
// Dense GEMM: C[M, Nout] = A[M, K] @ W[Nout, K]^T  (+ optional addsrc)
// A, W are f16. Block = 128 threads = 4 waves; block tile 64(M) x 128(N);
// wave tile 32x64 (2x4 WMMA frags, 8 wmma/K-step). Double-buffered async LDS.
// grid = (M/64, Nout/128). K % 32 == 0.
// ---------------------------------------------------------------------------
__global__ __launch_bounds__(128) void k_gemm(const _Float16* __restrict__ A,
                                              const _Float16* __restrict__ W,
                                              const float* __restrict__ addsrc,
                                              float* __restrict__ C,
                                              int Nout, int K) {
  __shared__ alignas(16) _Float16 As[2][64 * 32];
  __shared__ alignas(16) _Float16 Bs[2][128 * 32];
  const int bm = blockIdx.x * 64;
  const int bn = blockIdx.y * 128;
  const int tid = threadIdx.x;
  const int wave = tid >> 5, lane = tid & 31;
  const int wm = (wave & 1) * 32, wn = (wave >> 1) * 64;

  // Hoisted per-thread staging pointers (K-invariant part).
  const _Float16* aptr[2];
  int lofA[2];
#pragma unroll
  for (int i = 0; i < 2; ++i) {
    const int c = tid + 128 * i;          // 256 chunks: 64 rows x 4
    const int r = c >> 2, of = (c & 3) * 8;
    aptr[i] = A + (size_t)(bm + r) * K + of;
    lofA[i] = r * 32 + of;
  }
  const _Float16* wptr[4];
  int lofB[4];
#pragma unroll
  for (int i = 0; i < 4; ++i) {
    const int c = tid + 128 * i;          // 512 chunks: 128 rows x 4
    const int r = c >> 2, of = (c & 3) * 8;
    wptr[i] = W + (size_t)(bn + r) * K + of;
    lofB[i] = r * 32 + of;
  }

  v8f acc[2][4];
#pragma unroll
  for (int i = 0; i < 2; ++i)
#pragma unroll
    for (int j = 0; j < 4; ++j)
#pragma unroll
      for (int r = 0; r < 8; ++r) acc[i][j][r] = 0.0f;

  auto stage = [&](int b, int k0) {
#pragma unroll
    for (int i = 0; i < 2; ++i) cp16(aptr[i] + k0, &As[b][lofA[i]]);
#pragma unroll
    for (int i = 0; i < 4; ++i) cp16(wptr[i] + k0, &Bs[b][lofB[i]]);
  };
  auto compute = [&](int b) {
    const v16h a0 = frag_a(&As[b][0], wm, lane);
    const v16h a1 = frag_a(&As[b][0], wm + 16, lane);
#pragma unroll
    for (int j = 0; j < 4; ++j) {
      const v16h bj = frag_b(&Bs[b][0], wn + j * 16, lane);
      acc[0][j] = wmma16(a0, bj, acc[0][j]);
      acc[1][j] = wmma16(a1, bj, acc[1][j]);
    }
  };

  stage(0, 0);
  wait_async();
  __syncthreads();
  int buf = 0;
  for (int k0 = 32; k0 < K; k0 += 32) {
    stage(buf ^ 1, k0);   // async fill of the other buffer
    compute(buf);         // WMMA on current buffer
    wait_async();
    __syncthreads();
    buf ^= 1;
  }
  compute(buf);

  const int nn = lane & 15, hf = lane >> 4;
#pragma unroll
  for (int i = 0; i < 2; ++i)
#pragma unroll
    for (int j = 0; j < 4; ++j) {
      const int col = bn + wn + j * 16 + nn;
#pragma unroll
      for (int r = 0; r < 8; ++r) {
        const int row = bm + wm + i * 16 + hf * 8 + r;
        const size_t idx = (size_t)row * Nout + col;
        float v = acc[i][j][r];
        if (addsrc) v += addsrc[idx];
        C[idx] = v;
      }
    }
}

// ---------------------------------------------------------------------------
// MoE grouped/gather GEMM. Expert e = blockIdx.z owns compacted pair slots
// [offs[e], offs[e+1]). A-row = pairtok[slot] (gather) or slot itself.
// Edge rows are CLAMPED (not zeroed): garbage A rows only feed output rows
// whose stores are guarded, keeping staging straight-line with full EXEC.
// grid = (NPAIR/64, Nout/128, NEXP).
// ---------------------------------------------------------------------------
__global__ __launch_bounds__(128) void k_moe_gemm(const _Float16* __restrict__ A,
                                                  const int* __restrict__ pairtok,
                                                  const int* __restrict__ offs,
                                                  const _Float16* __restrict__ W,
                                                  float* __restrict__ C,
                                                  int Nout, int K) {
  const int e = blockIdx.z;
  const int start = offs[e], end = offs[e + 1];
  const int bm = start + blockIdx.x * 64;
  if (bm >= end) return;
  const int bn = blockIdx.y * 128;

  __shared__ alignas(16) _Float16 As[2][64 * 32];
  __shared__ alignas(16) _Float16 Bs[2][128 * 32];
  const int tid = threadIdx.x;
  const int wave = tid >> 5, lane = tid & 31;
  const int wm = (wave & 1) * 32, wn = (wave >> 1) * 64;
  const _Float16* We = W + (size_t)e * Nout * K;

  const _Float16* aptr[2];
  int lofA[2];
#pragma unroll
  for (int i = 0; i < 2; ++i) {
    const int c = tid + 128 * i;
    const int r = c >> 2, of = (c & 3) * 8;
    const int slot = min(bm + r, end - 1);
    const int arow = pairtok ? pairtok[slot] : slot;
    aptr[i] = A + (size_t)arow * K + of;
    lofA[i] = r * 32 + of;
  }
  const _Float16* wptr[4];
  int lofB[4];
#pragma unroll
  for (int i = 0; i < 4; ++i) {
    const int c = tid + 128 * i;
    const int r = c >> 2, of = (c & 3) * 8;
    wptr[i] = We + (size_t)(bn + r) * K + of;
    lofB[i] = r * 32 + of;
  }

  v8f acc[2][4];
#pragma unroll
  for (int i = 0; i < 2; ++i)
#pragma unroll
    for (int j = 0; j < 4; ++j)
#pragma unroll
      for (int r = 0; r < 8; ++r) acc[i][j][r] = 0.0f;

  auto stage = [&](int b, int k0) {
#pragma unroll
    for (int i = 0; i < 2; ++i) cp16(aptr[i] + k0, &As[b][lofA[i]]);
#pragma unroll
    for (int i = 0; i < 4; ++i) cp16(wptr[i] + k0, &Bs[b][lofB[i]]);
  };
  auto compute = [&](int b) {
    const v16h a0 = frag_a(&As[b][0], wm, lane);
    const v16h a1 = frag_a(&As[b][0], wm + 16, lane);
#pragma unroll
    for (int j = 0; j < 4; ++j) {
      const v16h bj = frag_b(&Bs[b][0], wn + j * 16, lane);
      acc[0][j] = wmma16(a0, bj, acc[0][j]);
      acc[1][j] = wmma16(a1, bj, acc[1][j]);
    }
  };

  stage(0, 0);
  wait_async();
  __syncthreads();
  int buf = 0;
  for (int k0 = 32; k0 < K; k0 += 32) {
    stage(buf ^ 1, k0);
    compute(buf);
    wait_async();
    __syncthreads();
    buf ^= 1;
  }
  compute(buf);

  const int nn = lane & 15, hf = lane >> 4;
#pragma unroll
  for (int i = 0; i < 2; ++i)
#pragma unroll
    for (int j = 0; j < 4; ++j) {
      const int col = bn + wn + j * 16 + nn;
#pragma unroll
      for (int r = 0; r < 8; ++r) {
        const int slot = bm + wm + i * 16 + hf * 8 + r;
        if (slot < end) C[(size_t)slot * Nout + col] = acc[i][j][r];
      }
    }
}

// ---------------------------------------------------------------------------
// Small VALU kernels
// ---------------------------------------------------------------------------
__global__ void k_cvt16(const float* __restrict__ s, _Float16* __restrict__ d,
                        int n4) {  // n4 = n/4
  const int i = blockIdx.x * 256 + threadIdx.x;
  if (i >= n4) return;
  const float4 f = *(const float4*)(s + (size_t)i * 4);
  _Float16* o = d + (size_t)i * 4;
  o[0] = (_Float16)f.x; o[1] = (_Float16)f.y;
  o[2] = (_Float16)f.z; o[3] = (_Float16)f.w;
}

__global__ void k_embed(const int* __restrict__ ids, const float* __restrict__ emb,
                        float* __restrict__ x) {
  const int idx = blockIdx.x * 256 + threadIdx.x;
  if (idx >= NTOK * D_MODEL) return;
  const int n = idx / D_MODEL, d = idx - n * D_MODEL;
  x[idx] = emb[(size_t)ids[n] * D_MODEL + d];
}

// RMSNorm: f32 in, f16 out (GEMM operand precision).
__global__ __launch_bounds__(256) void k_rmsnorm(const float* __restrict__ x,
                                                 const float* __restrict__ w,
                                                 _Float16* __restrict__ y) {
  __shared__ float red[256];
  const int row = blockIdx.x, tid = threadIdx.x;
  const float* xr = x + (size_t)row * D_MODEL;
  float s = 0.f;
  for (int d = tid; d < D_MODEL; d += 256) { const float v = xr[d]; s += v * v; }
  red[tid] = s;
  __syncthreads();
  for (int off = 128; off > 0; off >>= 1) {
    if (tid < off) red[tid] += red[tid + off];
    __syncthreads();
  }
  const float inv = 1.0f / sqrtf(red[0] / D_MODEL + 1e-6f);
  for (int d = tid; d < D_MODEL; d += 256)
    y[(size_t)row * D_MODEL + d] = (_Float16)(w[d] * xr[d] * inv);
}

__global__ void k_rope(float* __restrict__ buf, int nheads) {
  const int idx = blockIdx.x * 256 + threadIdx.x;
  const int total = NTOK * nheads * (HEADD / 2);
  if (idx >= total) return;
  const int i = idx & 31;
  const int h = (idx >> 5) % nheads;
  const int n = idx / (32 * nheads);
  const int t = n % SEQ;
  const float freq = __powf(10000.0f, -(float)i / 32.0f);
  float s, c;
  __sincosf((float)t * freq, &s, &c);
  float* p = buf + (size_t)n * (nheads * HEADD) + h * HEADD + 2 * i;
  const float xe = p[0], xo = p[1];
  p[0] = xe * c - xo * s;
  p[1] = xe * s + xo * c;
}

// Online-softmax causal attention (f32 math, f16 output for the o-proj GEMM).
__global__ __launch_bounds__(256) void k_attn(const float* __restrict__ qb,
                                              const float* __restrict__ kb,
                                              const float* __restrict__ vb,
                                              _Float16* __restrict__ ao) {
  __shared__ float Ks[64][64];
  __shared__ float Vs[64][64];
  const int tid = threadIdx.x;
  const int h = blockIdx.y, b = blockIdx.z;
  const int q0 = blockIdx.x * 256;
  const int q = q0 + tid;
  const int kvh = h / (NHEAD / NKVH);
  const size_t nq = (size_t)b * SEQ + q;

  float qreg[HEADD], acc[HEADD];
  const float* qr = qb + nq * QDIM + h * HEADD;
#pragma unroll
  for (int d = 0; d < HEADD; ++d) { qreg[d] = qr[d]; acc[d] = 0.f; }
  float m = -3.0e38f, l = 0.f;

  const int ktiles = q0 / 64 + 4;
  for (int kt = 0; kt < ktiles; ++kt) {
    for (int i = tid; i < 1024; i += 256) {
      const int r = i >> 4, c4 = (i & 15) << 2;
      const size_t nk = (size_t)b * SEQ + kt * 64 + r;
      *(float4*)&Ks[r][c4] = *(const float4*)(kb + nk * KVDIM + kvh * HEADD + c4);
      *(float4*)&Vs[r][c4] = *(const float4*)(vb + nk * KVDIM + kvh * HEADD + c4);
    }
    __syncthreads();
    const int kmax = min(64, q - kt * 64 + 1);
    for (int kk = 0; kk < kmax; ++kk) {
      float s = 0.f;
#pragma unroll
      for (int d = 0; d < HEADD; ++d) s += qreg[d] * Ks[kk][d];
      s *= 0.125f;
      const float mnew = fmaxf(m, s);
      const float corr = __expf(m - mnew);
      const float pv = __expf(s - mnew);
      l = l * corr + pv;
#pragma unroll
      for (int d = 0; d < HEADD; ++d) acc[d] = acc[d] * corr + pv * Vs[kk][d];
      m = mnew;
    }
    __syncthreads();
  }
  const float invl = 1.0f / l;
  _Float16* o = ao + nq * QDIM + h * HEADD;
#pragma unroll
  for (int d = 0; d < HEADD; ++d) o[d] = (_Float16)(acc[d] * invl);
}

__global__ void k_zero_counts(int* counts, int* fill) {
  const int t = threadIdx.x;
  if (t < NEXP) { counts[t] = 0; fill[t] = 0; }
}

__global__ void k_router(const _Float16* __restrict__ xn, const float* __restrict__ rw,
                         const float* __restrict__ rb, int* __restrict__ top_idx,
                         float* __restrict__ top_w, int* __restrict__ counts) {
  const int n = blockIdx.x * 256 + threadIdx.x;
  if (n >= NTOK) return;
  const _Float16* xr = xn + (size_t)n * D_MODEL;
  float lg[NEXP];
#pragma unroll
  for (int e = 0; e < NEXP; ++e) {
    const float* wr = rw + (size_t)e * D_MODEL;
    float s = 0.f;
    for (int d = 0; d < D_MODEL; ++d) s += (float)xr[d] * wr[d];
    lg[e] = s + rb[e];
  }
  float mx = lg[0];
#pragma unroll
  for (int e = 1; e < NEXP; ++e) mx = fmaxf(mx, lg[e]);
  float sum = 0.f;
#pragma unroll
  for (int e = 0; e < NEXP; ++e) { lg[e] = __expf(lg[e] - mx); sum += lg[e]; }
  const float inv = 1.0f / sum;
  int i0 = 0; float p0 = -1.f;
#pragma unroll
  for (int e = 0; e < NEXP; ++e) if (lg[e] > p0) { p0 = lg[e]; i0 = e; }
  int i1 = 0; float p1 = -1.f;
#pragma unroll
  for (int e = 0; e < NEXP; ++e) if (e != i0 && lg[e] > p1) { p1 = lg[e]; i1 = e; }
  p0 *= inv; p1 *= inv;
  const float wn = 1.0f / (p0 + p1 + 1e-9f);
  top_idx[2 * n] = i0; top_idx[2 * n + 1] = i1;
  top_w[2 * n] = p0 * wn; top_w[2 * n + 1] = p1 * wn;
  atomicAdd(&counts[i0], 1);
  atomicAdd(&counts[i1], 1);
}

__global__ void k_offsets(const int* __restrict__ counts, int* __restrict__ offs) {
  if (threadIdx.x == 0 && blockIdx.x == 0) {
    int a = 0;
    for (int e = 0; e < NEXP; ++e) { offs[e] = a; a += counts[e]; }
    offs[NEXP] = a;
  }
}

__global__ void k_fill(const int* __restrict__ top_idx, const float* __restrict__ top_w,
                       const int* __restrict__ offs, int* __restrict__ fill,
                       int* __restrict__ pairtok, float* __restrict__ pairw,
                       int* __restrict__ slots) {
  const int n = blockIdx.x * 256 + threadIdx.x;
  if (n >= NTOK) return;
#pragma unroll
  for (int j = 0; j < TOPK_; ++j) {
    const int e = top_idx[2 * n + j];
    const int slot = offs[e] + atomicAdd(&fill[e], 1);
    pairtok[slot] = n;
    pairw[slot] = top_w[2 * n + j];
    slots[2 * n + j] = slot;
  }
}

__global__ void k_silu_mul(const float* __restrict__ g, const float* __restrict__ u,
                           _Float16* __restrict__ h) {
  const int idx = blockIdx.x * 256 + threadIdx.x;
  if (idx >= NPAIR * HIDDEN) return;
  const float x = g[idx];
  h[idx] = (_Float16)(x * (1.0f / (1.0f + __expf(-x))) * u[idx]);
}

__global__ void k_scatter(const float* __restrict__ xb, const float* __restrict__ dp,
                          const int* __restrict__ slots, const float* __restrict__ pairw,
                          float* __restrict__ xo) {
  const int idx = blockIdx.x * 256 + threadIdx.x;
  if (idx >= NTOK * D_MODEL) return;
  const int n = idx / D_MODEL;
  const int d = idx - n * D_MODEL;
  const int s0 = slots[2 * n], s1 = slots[2 * n + 1];
  xo[idx] = xb[idx] + pairw[s0] * dp[(size_t)s0 * D_MODEL + d]
                    + pairw[s1] * dp[(size_t)s1 * D_MODEL + d];
}

// ---------------------------------------------------------------------------
// Host driver
// ---------------------------------------------------------------------------
extern "C" void kernel_launch(void* const* d_in, const int* in_sizes, int n_in,
                              void* d_out, int out_size, void* d_ws, size_t ws_size,
                              hipStream_t stream) {
  (void)in_sizes; (void)n_in; (void)out_size; (void)ws_size;
  const int*   ids   = (const int*)d_in[0];
  const float* emb   = (const float*)d_in[1];
  const float* anw   = (const float*)d_in[2];
  const float* qw    = (const float*)d_in[3];
  const float* kw    = (const float*)d_in[4];
  const float* vw    = (const float*)d_in[5];
  const float* ow    = (const float*)d_in[6];
  const float* fnw   = (const float*)d_in[7];
  const float* rw    = (const float*)d_in[8];
  const float* rb    = (const float*)d_in[9];
  const float* gw    = (const float*)d_in[10];
  const float* uw    = (const float*)d_in[11];
  const float* dw    = (const float*)d_in[12];
  const float* normw = (const float*)d_in[13];
  float* out = (float*)d_out;

  char* p = (char*)d_ws;
  auto carve = [&](size_t bytes) -> void* {
    void* r = (void*)p;
    p += (bytes + 255) & ~(size_t)255;
    return r;
  };
  float*    xa    = (float*)carve((size_t)NTOK * D_MODEL * 4);
  float*    xb    = (float*)carve((size_t)NTOK * D_MODEL * 4);
  _Float16* xn16  = (_Float16*)carve((size_t)NTOK * D_MODEL * 2);
  float*    qbuf  = (float*)carve((size_t)NTOK * QDIM * 4);
  float*    kbuf  = (float*)carve((size_t)NTOK * KVDIM * 4);
  float*    vbuf  = (float*)carve((size_t)NTOK * KVDIM * 4);
  _Float16* ao16  = (_Float16*)carve((size_t)NTOK * QDIM * 2);
  float*    gbuf  = (float*)carve((size_t)NPAIR * HIDDEN * 4);
  float*    ubuf  = (float*)carve((size_t)NPAIR * HIDDEN * 4);  // also down-out
  _Float16* h16   = (_Float16*)carve((size_t)NPAIR * HIDDEN * 2);
  float*    pairw = (float*)carve((size_t)NPAIR * 4);
  float*    topw  = (float*)carve((size_t)NTOK * TOPK_ * 4);
  int* topidx  = (int*)carve((size_t)NTOK * TOPK_ * 4);
  int* pairtok = (int*)carve((size_t)NPAIR * 4);
  int* slots   = (int*)carve((size_t)NTOK * TOPK_ * 4);
  int* counts  = (int*)carve(NEXP * 4);
  int* offs    = (int*)carve((NEXP + 1) * 4);
  int* fill    = (int*)carve(NEXP * 4);

  // One reusable f16 weight region (max of per-layer 27.8M and embed 32.8M elems)
  _Float16* w16 = (_Float16*)carve((size_t)VOCAB * D_MODEL * 2);
  _Float16* qw16 = w16;
  _Float16* kw16 = qw16 + (size_t)QDIM * D_MODEL;
  _Float16* vw16 = kw16 + (size_t)KVDIM * D_MODEL;
  _Float16* ow16 = vw16 + (size_t)KVDIM * D_MODEL;
  _Float16* gw16 = ow16 + (size_t)D_MODEL * QDIM;
  _Float16* uw16 = gw16 + (size_t)NEXP * HIDDEN * D_MODEL;
  _Float16* dw16 = uw16 + (size_t)NEXP * HIDDEN * D_MODEL;

  auto cvt = [&](const float* s, _Float16* d, size_t n) {
    k_cvt16<<<(int)(n / 1024), 256, 0, stream>>>(s, d, (int)(n / 4));
  };

  // Embedding gather
  k_embed<<<(NTOK * D_MODEL) / 256, 256, 0, stream>>>(ids, emb, xa);

  for (int l = 0; l < NLAYER; ++l) {
    // Per-layer weight conversion to f16 (re-done every call; deterministic)
    cvt(qw + (size_t)l * QDIM * D_MODEL,          qw16, (size_t)QDIM * D_MODEL);
    cvt(kw + (size_t)l * KVDIM * D_MODEL,         kw16, (size_t)KVDIM * D_MODEL);
    cvt(vw + (size_t)l * KVDIM * D_MODEL,         vw16, (size_t)KVDIM * D_MODEL);
    cvt(ow + (size_t)l * D_MODEL * QDIM,          ow16, (size_t)D_MODEL * QDIM);
    cvt(gw + (size_t)l * NEXP * HIDDEN * D_MODEL, gw16, (size_t)NEXP * HIDDEN * D_MODEL);
    cvt(uw + (size_t)l * NEXP * HIDDEN * D_MODEL, uw16, (size_t)NEXP * HIDDEN * D_MODEL);
    cvt(dw + (size_t)l * NEXP * D_MODEL * HIDDEN, dw16, (size_t)NEXP * D_MODEL * HIDDEN);

    // ---- attention block ----
    k_rmsnorm<<<NTOK, 256, 0, stream>>>(xa, anw + (size_t)l * D_MODEL, xn16);
    k_gemm<<<dim3(NTOK / 64, QDIM / 128), 128, 0, stream>>>(
        xn16, qw16, nullptr, qbuf, QDIM, D_MODEL);
    k_gemm<<<dim3(NTOK / 64, KVDIM / 128), 128, 0, stream>>>(
        xn16, kw16, nullptr, kbuf, KVDIM, D_MODEL);
    k_gemm<<<dim3(NTOK / 64, KVDIM / 128), 128, 0, stream>>>(
        xn16, vw16, nullptr, vbuf, KVDIM, D_MODEL);
    k_rope<<<(NTOK * NHEAD * 32) / 256, 256, 0, stream>>>(qbuf, NHEAD);
    k_rope<<<(NTOK * NKVH * 32) / 256, 256, 0, stream>>>(kbuf, NKVH);
    k_attn<<<dim3(SEQ / 256, NHEAD, BATCH), 256, 0, stream>>>(qbuf, kbuf, vbuf, ao16);
    k_gemm<<<dim3(NTOK / 64, D_MODEL / 128), 128, 0, stream>>>(
        ao16, ow16, xa, xb, D_MODEL, QDIM);  // residual fused

    // ---- MoE block (sparse top-2; same math as the dense reference) ----
    k_rmsnorm<<<NTOK, 256, 0, stream>>>(xb, fnw + (size_t)l * D_MODEL, xn16);
    k_zero_counts<<<1, 32, 0, stream>>>(counts, fill);
    k_router<<<NTOK / 256, 256, 0, stream>>>(
        xn16, rw + (size_t)l * NEXP * D_MODEL, rb + (size_t)l * NEXP,
        topidx, topw, counts);
    k_offsets<<<1, 1, 0, stream>>>(counts, offs);
    k_fill<<<NTOK / 256, 256, 0, stream>>>(topidx, topw, offs, fill,
                                           pairtok, pairw, slots);
    k_moe_gemm<<<dim3(NPAIR / 64, HIDDEN / 128, NEXP), 128, 0, stream>>>(
        xn16, pairtok, offs, gw16, gbuf, HIDDEN, D_MODEL);
    k_moe_gemm<<<dim3(NPAIR / 64, HIDDEN / 128, NEXP), 128, 0, stream>>>(
        xn16, pairtok, offs, uw16, ubuf, HIDDEN, D_MODEL);
    k_silu_mul<<<(NPAIR * HIDDEN) / 256, 256, 0, stream>>>(gbuf, ubuf, h16);
    k_moe_gemm<<<dim3(NPAIR / 64, D_MODEL / 128, NEXP), 128, 0, stream>>>(
        h16, nullptr, offs, dw16, ubuf, D_MODEL, HIDDEN);
    k_scatter<<<(NTOK * D_MODEL) / 256, 256, 0, stream>>>(xb, ubuf, slots, pairw, xa);
  }

  // ---- final norm + logits (2048 x 32000 x 1024, f16 embed fits in L2) ----
  k_rmsnorm<<<NTOK, 256, 0, stream>>>(xa, normw, xn16);
  cvt(emb, w16, (size_t)VOCAB * D_MODEL);
  k_gemm<<<dim3(NTOK / 64, VOCAB / 128), 128, 0, stream>>>(
      xn16, w16, nullptr, out, VOCAB, D_MODEL);
}
